// SimpleMultiheadAttention_16028817949526
// MI455X (gfx1250) — matrix operand order repaired and checked by reading, not compiled, not run on hardware
//
#include <hip/hip_runtime.h>
#include <hip/hip_bf16.h>

typedef __attribute__((ext_vector_type(16))) __bf16 v16bf;
typedef __attribute__((ext_vector_type(8)))  __bf16 v8bf;
typedef __attribute__((ext_vector_type(8)))  float  v8f;
typedef __attribute__((ext_vector_type(4)))  int    v4i;

union BF16x16 { v16bf v; v8bf h[2]; };

constexpr int BB = 2, T = 1024, E = 1024, H = 16, HD = 64;
constexpr int MROWS = BB * T;                 // 2048
constexpr float SCALE_TEMP = 0.25f;           // (1/sqrt(HD)) / TEMP = 0.125/0.5

#if __has_builtin(__builtin_amdgcn_global_load_async_to_lds_b128) && \
    __has_builtin(__builtin_amdgcn_s_wait_asynccnt)
#define HAVE_ASYNC 1
#else
#define HAVE_ASYNC 0
#endif

#define AS1 __attribute__((address_space(1)))
#define AS3 __attribute__((address_space(3)))

// Stage one 16-byte chunk global -> LDS. Async DMA path on gfx1250 toolchains
// that expose the builtin; falls back to reg round-trip otherwise.
__device__ __forceinline__ void stage16(const __bf16* g, __bf16* l) {
#if HAVE_ASYNC
    __builtin_amdgcn_global_load_async_to_lds_b128(
        (AS1 v4i*)(void*)g, (AS3 v4i*)(void*)l, 0, 0);
#else
    *(v8bf*)l = *(const v8bf*)g;
#endif
}
__device__ __forceinline__ void stage_wait() {
#if HAVE_ASYNC
    __builtin_amdgcn_s_wait_asynccnt(0);
#endif
}

__device__ __forceinline__ v8f wmma_bf16(const BF16x16& a, const BF16x16& b, v8f c) {
    return __builtin_amdgcn_wmma_f32_16x16x32_bf16(false, a.v, false, b.v,
                                                   (short)0, c, false, false);
}

// ---------------------------------------------------------------- f32 -> bf16
__global__ void cvt_f32_bf16(const float* __restrict__ src,
                             __bf16* __restrict__ dst, int n) {
    int i = blockIdx.x * blockDim.x + threadIdx.x;
    int stride = gridDim.x * blockDim.x;
    for (; i < n; i += stride) dst[i] = (__bf16)src[i];
}

// ------------------------------------------------------- C = A * W^T + bias
// A: [M x K] bf16 row-major, W: [N x K] bf16 row-major, bias: [N] f32
// mode 0: f32 out [M x N]
// mode 1: bf16 out as heads [B,H,T,HD]
// mode 2: bf16 out as transposed heads [B,H,HD,T]   (for V)
// Double-buffered LDS: async-stage tile k+1 while computing tile k.
__global__ __launch_bounds__(256)
void gemm_bias_wmma(const __bf16* __restrict__ A, const __bf16* __restrict__ W,
                    const float* __restrict__ bias, void* __restrict__ out,
                    int M, int N, int K, int mode) {
    __shared__ __align__(16) __bf16 lds_a[2][128 * 40];
    __shared__ __align__(16) __bf16 lds_w[2][64 * 40];
    const int tid = threadIdx.x;
    const int wave = tid >> 5;
    const int lane = tid & 31;
    const int half = lane >> 4;
    const int l16 = lane & 15;
    const int bm = blockIdx.y;   // M / 128
    const int bn = blockIdx.x;   // N / 64

    auto stage = [&](int buf, int k0) {
        #pragma unroll
        for (int c = 0; c < 2; ++c) {          // A tile 128x32: 512 x 16B chunks
            int i = tid + c * 256;
            int row = i >> 2, ch = i & 3;
            stage16(&A[(size_t)(bm * 128 + row) * K + k0 + ch * 8],
                    &lds_a[buf][row * 40 + ch * 8]);
        }
        {                                       // W tile 64x32: 256 x 16B chunks
            int row = tid >> 2, ch = tid & 3;
            stage16(&W[(size_t)(bn * 64 + row) * K + k0 + ch * 8],
                    &lds_w[buf][row * 40 + ch * 8]);
        }
    };

    v8f acc[4] = {};
    stage(0, 0);
    stage_wait();
    __syncthreads();

    const int kiters = K >> 5;
    for (int it = 0; it < kiters; ++it) {
        const int cur = it & 1, nxt = cur ^ 1;
        if (it + 1 < kiters) stage(nxt, (it + 1) * 32);   // prefetch next tile

        BF16x16 af;
        const int arow = wave * 16 + l16;
        af.h[0] = *(const v8bf*)&lds_a[cur][arow * 40 + half * 8];
        af.h[1] = *(const v8bf*)&lds_a[cur][arow * 40 + 16 + half * 8];
        #pragma unroll
        for (int nt = 0; nt < 4; ++nt) {
            BF16x16 bf;
            const int wrow = nt * 16 + l16;
            bf.h[0] = *(const v8bf*)&lds_w[cur][wrow * 40 + half * 8];
            bf.h[1] = *(const v8bf*)&lds_w[cur][wrow * 40 + 16 + half * 8];
            acc[nt] = wmma_bf16(af, bf, acc[nt]);
        }
        stage_wait();
        __syncthreads();
    }

    const int mbase = bm * 128 + wave * 16 + half * 8;
    #pragma unroll
    for (int nt = 0; nt < 4; ++nt) {
        const int n = bn * 64 + nt * 16 + l16;
        const float bvv = bias[n];
        #pragma unroll
        for (int r = 0; r < 8; ++r) {
            const int m = mbase + r;
            const float vout = acc[nt][r] + bvv;
            if (mode == 0) {
                ((float*)out)[(size_t)m * N + n] = vout;
            } else {
                const int bi = m >> 10, t = m & (T - 1);
                const int h = n >> 6, d = n & (HD - 1);
                if (mode == 1)
                    ((__bf16*)out)[(((size_t)(bi * H + h)) * T + t) * HD + d] = (__bf16)vout;
                else
                    ((__bf16*)out)[(((size_t)(bi * H + h)) * HD + d) * T + t] = (__bf16)vout;
            }
        }
    }
}

// ------------------------- pass 1: Z_i = sum exp(S_i/TEMP) per (b,h), i=1..3
// grid (64 tiles, 32 bh), block 256. Each block: 128x128 score region.
__global__ __launch_bounds__(256)
void attn_stats(const __bf16* __restrict__ Q1, const __bf16* __restrict__ Q2,
                const __bf16* __restrict__ Q3, const __bf16* __restrict__ Kh,
                float* __restrict__ partials) {
    __shared__ __align__(16) __bf16 lds_k[128 * 72];
    __shared__ float red[8 * 3];
    const int tid = threadIdx.x, wave = tid >> 5, lane = tid & 31;
    const int half = lane >> 4, l16 = lane & 15;
    const int tile = blockIdx.x;            // 0..63
    const int bh = blockIdx.y;              // 0..31
    const int tb = tile >> 3, sb = tile & 7;

    // stage K block: 128 s-rows x 64 d
    #pragma unroll
    for (int c = 0; c < 4; ++c) {
        int i = tid + c * 256;
        int row = i >> 3, ch = i & 7;
        stage16(&Kh[(((size_t)bh) * T + sb * 128 + row) * HD + ch * 8],
                &lds_k[row * 72 + ch * 8]);
    }
    stage_wait();
    __syncthreads();

    const __bf16* Qs[3] = {Q1, Q2, Q3};
    BF16x16 qf[3][2];
    const int trow = tb * 128 + wave * 16 + l16;
    #pragma unroll
    for (int q = 0; q < 3; ++q) {
        const __bf16* base = &Qs[q][(((size_t)bh) * T + trow) * HD];
        #pragma unroll
        for (int f = 0; f < 2; ++f) {
            qf[q][f].h[0] = *(const v8bf*)&base[f * 32 + half * 8];
            qf[q][f].h[1] = *(const v8bf*)&base[f * 32 + 16 + half * 8];
        }
    }

    float lsum[3] = {0.f, 0.f, 0.f};
    for (int j = 0; j < 8; ++j) {
        BF16x16 kf[2];
        const int srow = j * 16 + l16;
        #pragma unroll
        for (int f = 0; f < 2; ++f) {
            kf[f].h[0] = *(const v8bf*)&lds_k[srow * 72 + f * 32 + half * 8];
            kf[f].h[1] = *(const v8bf*)&lds_k[srow * 72 + f * 32 + 16 + half * 8];
        }
        #pragma unroll
        for (int q = 0; q < 3; ++q) {
            v8f s = {};
            s = wmma_bf16(qf[q][0], kf[0], s);
            s = wmma_bf16(qf[q][1], kf[1], s);
            float a = 0.f;
            #pragma unroll
            for (int r = 0; r < 8; ++r) a += __expf(s[r] * SCALE_TEMP);
            lsum[q] += a;
        }
    }
    #pragma unroll
    for (int q = 0; q < 3; ++q) {
        float v = lsum[q];
        for (int m = 16; m >= 1; m >>= 1) v += __shfl_xor(v, m, 32);
        if (lane == 0) red[wave * 3 + q] = v;
    }
    __syncthreads();
    if (tid < 3) {
        float s = 0.f;
        for (int w = 0; w < 8; ++w) s += red[w * 3 + tid];
        partials[(((size_t)bh) * 64 + tile) * 3 + tid] = s;
    }
}

__global__ void reduce_z(const float* __restrict__ partials, float* __restrict__ coef) {
    const int bh = blockIdx.x;
    if (threadIdx.x < 3) {
        float s = 0.f;
        for (int t = 0; t < 64; ++t)
            s += partials[(((size_t)bh) * 64 + t) * 3 + threadIdx.x];
        coef[bh * 3 + threadIdx.x] = (float)T / s;   // a_i = coef_i * exp(x_i)
    }
}

// -------- pass 2: write a1 (f32) and attn_out bf16 = ((a1+a2+a3)/3) @ V
// grid (8 t-blocks, 32 bh), block 256 (8 waves x 16 rows = 128 t-rows/block)
__global__ __launch_bounds__(256)
void attn_pass2(const __bf16* __restrict__ Q1, const __bf16* __restrict__ Q2,
                const __bf16* __restrict__ Q3, const __bf16* __restrict__ Kh,
                const __bf16* __restrict__ Vt, const float* __restrict__ coef,
                float* __restrict__ a1_out, __bf16* __restrict__ attn_out) {
    __shared__ __align__(16) __bf16 lds_k[32 * 72];
    __shared__ __align__(16) __bf16 lds_v[64 * 40];
    __shared__ __align__(16) __bf16 lds_p[8 * 16 * 40];
    const int tid = threadIdx.x, wave = tid >> 5, lane = tid & 31;
    const int half = lane >> 4, l16 = lane & 15;
    const int tb = blockIdx.x;              // 0..7
    const int bh = blockIdx.y;              // 0..31
    const float c1 = coef[bh * 3 + 0];
    const float c2 = coef[bh * 3 + 1];
    const float c3 = coef[bh * 3 + 2];
    const float third = 1.0f / 3.0f;

    const __bf16* Qs[3] = {Q1, Q2, Q3};
    BF16x16 qf[3][2];
    const int trow = tb * 128 + wave * 16 + l16;
    #pragma unroll
    for (int q = 0; q < 3; ++q) {
        const __bf16* base = &Qs[q][(((size_t)bh) * T + trow) * HD];
        #pragma unroll
        for (int f = 0; f < 2; ++f) {
            qf[q][f].h[0] = *(const v8bf*)&base[f * 32 + half * 8];
            qf[q][f].h[1] = *(const v8bf*)&base[f * 32 + 16 + half * 8];
        }
    }

    v8f acc[4] = {};
    float* a1base = a1_out + ((size_t)bh) * T * T;
    const int tout = tb * 128 + wave * 16 + half * 8;

    for (int sc = 0; sc < T / 32; ++sc) {
        {   // stage K chunk: 32 s-rows x 64 d
            int row = tid >> 3, ch = tid & 7;
            stage16(&Kh[(((size_t)bh) * T + sc * 32 + row) * HD + ch * 8],
                    &lds_k[row * 72 + ch * 8]);
        }
        {   // stage Vt chunk: 64 d-rows x 32 s
            int row = tid >> 2, ch = tid & 3;
            stage16(&Vt[(((size_t)bh) * HD + row) * T + sc * 32 + ch * 8],
                    &lds_v[row * 40 + ch * 8]);
        }
        stage_wait();
        __syncthreads();

        #pragma unroll
        for (int j = 0; j < 2; ++j) {
            BF16x16 kf[2];
            const int srow = j * 16 + l16;
            #pragma unroll
            for (int f = 0; f < 2; ++f) {
                kf[f].h[0] = *(const v8bf*)&lds_k[srow * 72 + f * 32 + half * 8];
                kf[f].h[1] = *(const v8bf*)&lds_k[srow * 72 + f * 32 + 16 + half * 8];
            }
            v8f s1 = {}, s2 = {}, s3 = {};
            s1 = wmma_bf16(qf[0][0], kf[0], s1); s1 = wmma_bf16(qf[0][1], kf[1], s1);
            s2 = wmma_bf16(qf[1][0], kf[0], s2); s2 = wmma_bf16(qf[1][1], kf[1], s2);
            s3 = wmma_bf16(qf[2][0], kf[0], s3); s3 = wmma_bf16(qf[2][1], kf[1], s3);
            const int scol = sc * 32 + j * 16 + l16;
            #pragma unroll
            for (int r = 0; r < 8; ++r) {
                const float e1 = __expf(s1[r] * SCALE_TEMP);
                const float e2 = __expf(s2[r] * SCALE_TEMP);
                const float e3 = __expf(s3[r] * SCALE_TEMP);
                const float a1v = c1 * e1;
                a1base[(size_t)(tout + r) * T + scol] = a1v;                 // a1 output
                const float p = (a1v + c2 * e2 + c3 * e3) * third;
                lds_p[wave * 640 + (half * 8 + r) * 40 + j * 16 + l16] = (__bf16)p;
            }
        }
        __syncthreads();   // P tiles visible; lds_k reads done

        BF16x16 pf;
        pf.h[0] = *(const v8bf*)&lds_p[wave * 640 + l16 * 40 + half * 8];
        pf.h[1] = *(const v8bf*)&lds_p[wave * 640 + l16 * 40 + 16 + half * 8];
        #pragma unroll
        for (int dn = 0; dn < 4; ++dn) {
            BF16x16 vf;
            const int drow = dn * 16 + l16;
            vf.h[0] = *(const v8bf*)&lds_v[drow * 40 + half * 8];
            vf.h[1] = *(const v8bf*)&lds_v[drow * 40 + 16 + half * 8];
            acc[dn] = wmma_bf16(pf, vf, acc[dn]);
        }
        __syncthreads();   // before restaging lds_k / lds_v
    }

    const int bi = bh >> 4, h = bh & (H - 1);
    #pragma unroll
    for (int dn = 0; dn < 4; ++dn) {
        const int d = dn * 16 + l16;
        #pragma unroll
        for (int r = 0; r < 8; ++r) {
            const int t = tout + r;
            attn_out[((size_t)bi * T + t) * E + h * HD + d] = (__bf16)acc[dn][r];
        }
    }
}

// ----------------------------------------------------------------- launcher
extern "C" void kernel_launch(void* const* d_in, const int* in_sizes, int n_in,
                              void* d_out, int out_size, void* d_ws, size_t ws_size,
                              hipStream_t stream) {
    const float* q1f = (const float*)d_in[0];
    const float* q2f = (const float*)d_in[1];
    const float* kyf = (const float*)d_in[2];
    const float* vlf = (const float*)d_in[3];
    const float* Wq = (const float*)d_in[4];  const float* bq = (const float*)d_in[5];
    const float* Wq2 = (const float*)d_in[6]; const float* bq2 = (const float*)d_in[7];
    const float* Wq3 = (const float*)d_in[8]; const float* bq3 = (const float*)d_in[9];
    const float* Wk = (const float*)d_in[10]; const float* bk = (const float*)d_in[11];
    const float* Wv = (const float*)d_in[12]; const float* bv = (const float*)d_in[13];
    const float* Wo = (const float*)d_in[14]; const float* bo = (const float*)d_in[15];

    char* ws = (char*)d_ws;
    size_t off = 0;
    auto alloc = [&](size_t bytes) -> void* {
        void* p = ws + off;
        off += (bytes + 255) & ~(size_t)255;
        return p;
    };
    const size_t ACT = (size_t)MROWS * E * sizeof(__bf16);   // 4 MB
    const size_t WT = (size_t)E * E * sizeof(__bf16);        // 2 MB
    __bf16* q1b = (__bf16*)alloc(ACT);
    __bf16* q2b = (__bf16*)alloc(ACT);
    __bf16* kyb = (__bf16*)alloc(ACT);
    __bf16* vlb = (__bf16*)alloc(ACT);
    __bf16* Wqb = (__bf16*)alloc(WT);
    __bf16* Wq2b = (__bf16*)alloc(WT);
    __bf16* Wq3b = (__bf16*)alloc(WT);
    __bf16* Wkb = (__bf16*)alloc(WT);
    __bf16* Wvb = (__bf16*)alloc(WT);
    __bf16* Wob = (__bf16*)alloc(WT);
    __bf16* Q1h = (__bf16*)alloc(ACT);   // [B,H,T,HD]
    __bf16* Q2h = (__bf16*)alloc(ACT);
    __bf16* Q3h = (__bf16*)alloc(ACT);
    __bf16* Kh  = (__bf16*)alloc(ACT);
    __bf16* Vt  = (__bf16*)alloc(ACT);   // [B,H,HD,T]
    __bf16* AOb = (__bf16*)alloc(ACT);   // attn_out bf16 [B,T,E]
    float* Zpart = (float*)alloc((size_t)32 * 64 * 3 * sizeof(float));
    float* Zcoef = (float*)alloc((size_t)32 * 3 * sizeof(float));

    float* outp = (float*)d_out;                          // [B,T,E]
    float* a1p = outp + (size_t)MROWS * E;                // [B,H,T,T]

    const int actN = MROWS * E, wN = E * E;
    cvt_f32_bf16<<<1024, 256, 0, stream>>>(q1f, q1b, actN);
    cvt_f32_bf16<<<1024, 256, 0, stream>>>(q2f, q2b, actN);
    cvt_f32_bf16<<<1024, 256, 0, stream>>>(kyf, kyb, actN);
    cvt_f32_bf16<<<1024, 256, 0, stream>>>(vlf, vlb, actN);
    cvt_f32_bf16<<<512, 256, 0, stream>>>(Wq, Wqb, wN);
    cvt_f32_bf16<<<512, 256, 0, stream>>>(Wq2, Wq2b, wN);
    cvt_f32_bf16<<<512, 256, 0, stream>>>(Wq3, Wq3b, wN);
    cvt_f32_bf16<<<512, 256, 0, stream>>>(Wk, Wkb, wN);
    cvt_f32_bf16<<<512, 256, 0, stream>>>(Wv, Wvb, wN);
    cvt_f32_bf16<<<512, 256, 0, stream>>>(Wo, Wob, wN);

    dim3 blk(256);
    dim3 gG(E / 64, MROWS / 128);   // (16,16)
    gemm_bias_wmma<<<gG, blk, 0, stream>>>(q1b, Wqb, bq, Q1h, MROWS, E, E, 1);
    gemm_bias_wmma<<<gG, blk, 0, stream>>>(q2b, Wq2b, bq2, Q2h, MROWS, E, E, 1);
    gemm_bias_wmma<<<gG, blk, 0, stream>>>(kyb, Wq3b, bq3, Q3h, MROWS, E, E, 1);
    gemm_bias_wmma<<<gG, blk, 0, stream>>>(kyb, Wkb, bk, Kh, MROWS, E, E, 1);
    gemm_bias_wmma<<<gG, blk, 0, stream>>>(vlb, Wvb, bv, Vt, MROWS, E, E, 2);

    attn_stats<<<dim3(64, 32), blk, 0, stream>>>(Q1h, Q2h, Q3h, Kh, Zpart);
    reduce_z<<<32, 32, 0, stream>>>(Zpart, Zcoef);
    attn_pass2<<<dim3(8, 32), blk, 0, stream>>>(Q1h, Q2h, Q3h, Kh, Vt, Zcoef, a1p, AOb);

    gemm_bias_wmma<<<gG, blk, 0, stream>>>(AOb, Wob, bo, outp, MROWS, E, E, 0);
}
